// GraphAttnBlock_41532333753003
// MI455X (gfx1250) — matrix operand order repaired and checked by reading, not compiled, not run on hardware
//
#include <hip/hip_runtime.h>
#include <math.h>

#define N_NODES 50000
#define N_EDGES 800000
#define DHID    128   // H*C = 4*32

typedef __attribute__((ext_vector_type(16))) _Float16 v16h;
typedef __attribute__((ext_vector_type(8)))  float    v8f;

__device__ __forceinline__ void atomicMaxF(float* addr, float v) {
  if (v >= 0.0f) atomicMax((int*)addr, __float_as_int(v));
  else           atomicMin((unsigned int*)addr, __float_as_uint(v));
}

// ---------------------------------------------------------------------------
// GEMM: out[n,:] = X[n,:] @ W (+ biases for the lin/skip matrix).
// Split-precision f16 WMMA: W = B1 + B2, X = A1 + A2 (value + residual),
// acc += A1*B1 + A1*B2 + A2*B1  (error ~2^-22, near-f32).
// B stored in LDS pre-swizzled into the v_wmma_f32_16x16x32_f16 B-fragment
// layout so each lane reads a contiguous 32B fragment.
// ---------------------------------------------------------------------------
__global__ __launch_bounds__(128) void k_gemm3(
    const float* __restrict__ X,
    const float* __restrict__ Wsrc, const float* __restrict__ Wdst,
    const float* __restrict__ Wlin,
    const float* __restrict__ bLin, const float* __restrict__ bConv,
    float* __restrict__ outS, float* __restrict__ outD,
    float* __restrict__ outL)
{
  __shared__ _Float16 sB1[16384];   // 32 KB
  __shared__ _Float16 sB2[16384];   // 32 KB

  const int y = blockIdx.y;
  const float* W = (y == 0) ? Wsrc : (y == 1) ? Wdst : Wlin;
  float* outp    = (y == 0) ? outS : (y == 1) ? outD : outL;
  const bool addb = (y == 2);

  const int tid = threadIdx.x;

  // Coalesced read of W, scattered swizzle write into fragment layout:
  // B frag: lane = hi*16 + l (l = column%16, hi selects K half),
  //         element e -> K = kstep*32 + hi*16 + e, column = ntile*16 + l.
  for (int i = tid; i < 128 * 128; i += 128) {
    const int k = i >> 7, col = i & 127;
    const float w = W[i];
    const _Float16 w1 = (_Float16)w;
    const _Float16 w2 = (_Float16)(w - (float)w1);
    const int ntile = col >> 4, l = col & 15;
    const int kstep = k >> 5, within = k & 31;
    const int hi = within >> 4, e = within & 15;
    const int idx = (((ntile * 4 + kstep) * 32) + (hi * 16 + l)) * 16 + e;
    sB1[idx] = w1;
    sB2[idx] = w2;
  }
  __syncthreads();

  const int wave = tid >> 5;
  const int lane = tid & 31;
  const int l  = lane & 15;
  const int hi = lane >> 4;
  const int nblk = N_NODES / 16;  // 3125 exactly

  for (int blk = blockIdx.x * 4 + wave; blk < nblk; blk += gridDim.x * 4) {
    const int m0 = blk * 16;
    const int r  = m0 + l;

    // A fragments (contiguous k order, same as B): lane row = m0+l;
    // element e -> K = k0 + 16*hi + e  => one contiguous run of 16.
    v16h a1[4], a2[4];
#pragma unroll
    for (int ks = 0; ks < 4; ++ks) {
      const float* xp = X + (size_t)r * 128 + ks * 32 + hi * 16;
#pragma unroll
      for (int j = 0; j < 8; ++j) {
        float v = xp[j];
        _Float16 h1 = (_Float16)v;
        a1[ks][j]     = h1;
        a2[ks][j]     = (_Float16)(v - (float)h1);
        v  = xp[8 + j];
        h1 = (_Float16)v;
        a1[ks][8 + j] = h1;
        a2[ks][8 + j] = (_Float16)(v - (float)h1);
      }
    }

#pragma unroll
    for (int nt = 0; nt < 8; ++nt) {
      v8f c = {0.f, 0.f, 0.f, 0.f, 0.f, 0.f, 0.f, 0.f};
#pragma unroll
      for (int ks = 0; ks < 4; ++ks) {
        const v16h b1v = *(const v16h*)&sB1[(((nt * 4 + ks) * 32) + lane) * 16];
        const v16h b2v = *(const v16h*)&sB2[(((nt * 4 + ks) * 32) + lane) * 16];
        c = __builtin_amdgcn_wmma_f32_16x16x32_f16(false, a1[ks], false, b1v, (short)0, c, false, false);
        c = __builtin_amdgcn_wmma_f32_16x16x32_f16(false, a1[ks], false, b2v, (short)0, c, false, false);
        c = __builtin_amdgcn_wmma_f32_16x16x32_f16(false, a2[ks], false, b1v, (short)0, c, false, false);
      }
      const int col = nt * 16 + l;
      const float bb = addb ? (bLin[col] + bConv[col]) : 0.0f;
      // D layout: VGPR v -> row m0 + v + 8*hi, column = col (per lane)
#pragma unroll
      for (int v = 0; v < 8; ++v) {
        outp[(size_t)(m0 + v + 8 * hi) * 128 + col] = c[v] + bb;
      }
    }
  }
}

// ---------------------------------------------------------------------------
// Per-node attention logits: a_s[n,h] = sum_c xs[n,h*32+c]*attS[h*32+c], same
// for a_d. One wave per row, 4 columns per lane, shfl_xor reduce per head.
// ---------------------------------------------------------------------------
__global__ __launch_bounds__(256) void k_rowdot(
    const float* __restrict__ xs, const float* __restrict__ xd,
    const float* __restrict__ attS, const float* __restrict__ attD,
    float* __restrict__ a_s, float* __restrict__ a_d)
{
  const int row = blockIdx.x * 8 + (threadIdx.x >> 5);
  if (row >= N_NODES) return;
  const int t = threadIdx.x & 31;
  const int c0 = t * 4;
  const float4 xv = *(const float4*)&xs[(size_t)row * 128 + c0];
  const float4 av = *(const float4*)&attS[c0];
  const float4 yv = *(const float4*)&xd[(size_t)row * 128 + c0];
  const float4 bv = *(const float4*)&attD[c0];
  float p = xv.x * av.x + xv.y * av.y + xv.z * av.z + xv.w * av.w;
  float q = yv.x * bv.x + yv.y * bv.y + yv.z * bv.z + yv.w * bv.w;
#pragma unroll
  for (int m = 1; m <= 4; m <<= 1) {
    p += __shfl_xor(p, m, 32);
    q += __shfl_xor(q, m, 32);
  }
  if ((t & 7) == 0) {
    a_s[row * 4 + (t >> 3)] = p;
    a_d[row * 4 + (t >> 3)] = q;
  }
}

__global__ __launch_bounds__(256) void k_init_md(float* __restrict__ m,
                                                 float* __restrict__ den, int n)
{
  const int i = blockIdx.x * blockDim.x + threadIdx.x;
  if (i < n) {
    m[i] = __int_as_float(0xff800000);  // -inf
    den[i] = 0.0f;
  }
}

// Edge logits: leaky_relu(a_s[src] + a_d[dst]); atomic segment-max over dst.
__global__ __launch_bounds__(256) void k_edge_logit(
    const int* __restrict__ src, const int* __restrict__ dst,
    const float* __restrict__ a_s, const float* __restrict__ a_d,
    float* __restrict__ ebuf, float* __restrict__ mbuf)
{
  const int idx = blockIdx.x * blockDim.x + threadIdx.x;
  if (idx >= N_EDGES * 4) return;
  const int e = idx >> 2, h = idx & 3;
  const int s = src[e], d = dst[e];
  float v = a_s[s * 4 + h] + a_d[d * 4 + h];
  v = (v > 0.0f) ? v : 0.2f * v;
  ebuf[idx] = v;
  atomicMaxF(&mbuf[d * 4 + h], v);
}

// ee = exp(e - m[dst]); denom = segment_sum(ee)
__global__ __launch_bounds__(256) void k_edge_exp(
    const int* __restrict__ dst,
    float* __restrict__ ebuf, const float* __restrict__ mbuf,
    float* __restrict__ den)
{
  const int idx = blockIdx.x * blockDim.x + threadIdx.x;
  if (idx >= N_EDGES * 4) return;
  const int e = idx >> 2, h = idx & 3;
  const int d = dst[e];
  const float v = expf(ebuf[idx] - mbuf[d * 4 + h]);
  ebuf[idx] = v;
  atomicAdd(&den[d * 4 + h], v);
}

// out[dst] += xs[src] * alpha ; 32 threads per edge, float4 per thread
__global__ __launch_bounds__(256) void k_edge_aggr(
    const int* __restrict__ src, const int* __restrict__ dst,
    const float* __restrict__ ebuf, const float* __restrict__ den,
    const float* __restrict__ xs, float* __restrict__ acc)
{
  const int e = blockIdx.x * 8 + (threadIdx.x >> 5);
  if (e >= N_EDGES) return;
  const int t = threadIdx.x & 31;
  const int h = t >> 3;
  const int s = src[e], d = dst[e];
  const float alpha = ebuf[e * 4 + h] / (den[d * 4 + h] + 1e-16f);
  const float4 xv = *(const float4*)&xs[(size_t)s * 128 + t * 4];
  float* outp = &acc[(size_t)d * 128 + t * 4];
  atomicAdd(outp + 0, xv.x * alpha);
  atomicAdd(outp + 1, xv.y * alpha);
  atomicAdd(outp + 2, xv.z * alpha);
  atomicAdd(outp + 3, xv.w * alpha);
}

// LayerNorm over 128 + ReLU, one wave per row (population variance).
__global__ __launch_bounds__(256) void k_ln_relu(
    const float* __restrict__ acc, const float* __restrict__ gamma,
    const float* __restrict__ beta, float* __restrict__ hout)
{
  const int row = blockIdx.x * 8 + (threadIdx.x >> 5);
  if (row >= N_NODES) return;
  const int t = threadIdx.x & 31;
  const int c0 = t * 4;
  const float4 v = *(const float4*)&acc[(size_t)row * 128 + c0];
  float s  = v.x + v.y + v.z + v.w;
  float ss = v.x * v.x + v.y * v.y + v.z * v.z + v.w * v.w;
#pragma unroll
  for (int m = 1; m <= 16; m <<= 1) {
    s  += __shfl_xor(s, m, 32);
    ss += __shfl_xor(ss, m, 32);
  }
  const float mu  = s * (1.0f / 128.0f);
  const float var = ss * (1.0f / 128.0f) - mu * mu;
  const float inv = rsqrtf(var + 1e-5f);
  const float4 g = *(const float4*)&gamma[c0];
  const float4 b = *(const float4*)&beta[c0];
  float4 o;
  o.x = fmaxf((v.x - mu) * inv * g.x + b.x, 0.0f);
  o.y = fmaxf((v.y - mu) * inv * g.y + b.y, 0.0f);
  o.z = fmaxf((v.z - mu) * inv * g.z + b.z, 0.0f);
  o.w = fmaxf((v.w - mu) * inv * g.w + b.w, 0.0f);
  *(float4*)&hout[(size_t)row * 128 + c0] = o;
}

// ---------------------------------------------------------------------------
extern "C" void kernel_launch(void* const* d_in, const int* in_sizes, int n_in,
                              void* d_out, int out_size, void* d_ws, size_t ws_size,
                              hipStream_t stream) {
  const float* x    = (const float*)d_in[0];
  const int*   eidx = (const int*)d_in[1];
  const int*   src  = eidx;
  const int*   dst  = eidx + N_EDGES;
  const float* W1s  = (const float*)d_in[2];
  const float* W1d  = (const float*)d_in[3];
  const float* at1s = (const float*)d_in[4];
  const float* at1d = (const float*)d_in[5];
  const float* b1   = (const float*)d_in[6];
  const float* Wl1  = (const float*)d_in[7];
  const float* bl1  = (const float*)d_in[8];
  const float* gam  = (const float*)d_in[9];
  const float* bet  = (const float*)d_in[10];
  const float* W2s  = (const float*)d_in[11];
  const float* W2d  = (const float*)d_in[12];
  const float* at2s = (const float*)d_in[13];
  const float* at2d = (const float*)d_in[14];
  const float* b2   = (const float*)d_in[15];
  const float* Wl2  = (const float*)d_in[16];
  const float* bl2  = (const float*)d_in[17];
  float* out = (float*)d_out;

  // workspace carve (floats): 4 x N*128 + 4 x N*4 + E*4  ~= 118.4 MB
  float* f    = (float*)d_ws;
  float* xs   = f;
  float* xd   = xs  + (size_t)N_NODES * 128;
  float* acc  = xd  + (size_t)N_NODES * 128;
  float* hbuf = acc + (size_t)N_NODES * 128;
  float* a_s  = hbuf + (size_t)N_NODES * 128;
  float* a_d  = a_s + (size_t)N_NODES * 4;
  float* mbuf = a_d + (size_t)N_NODES * 4;
  float* den  = mbuf + (size_t)N_NODES * 4;
  float* ebuf = den + (size_t)N_NODES * 4;

  const int NH4 = N_NODES * 4;                 // 200000
  const int g_init = (NH4 + 255) / 256;        // 782
  const int g_row  = (N_NODES + 7) / 8;        // 6250
  const int g_eh   = (N_EDGES * 4 + 255) / 256;// 12500
  const int g_agg  = (N_EDGES + 7) / 8;        // 100000
  const dim3 g_gemm(120, 3);

  // ---- layer 1 ----
  k_init_md<<<g_init, 256, 0, stream>>>(mbuf, den, NH4);
  k_gemm3<<<g_gemm, 128, 0, stream>>>(x, W1s, W1d, Wl1, bl1, b1, xs, xd, acc);
  k_rowdot<<<g_row, 256, 0, stream>>>(xs, xd, at1s, at1d, a_s, a_d);
  k_edge_logit<<<g_eh, 256, 0, stream>>>(src, dst, a_s, a_d, ebuf, mbuf);
  k_edge_exp<<<g_eh, 256, 0, stream>>>(dst, ebuf, mbuf, den);
  k_edge_aggr<<<g_agg, 256, 0, stream>>>(src, dst, ebuf, den, xs, acc);
  k_ln_relu<<<g_row, 256, 0, stream>>>(acc, gam, bet, hbuf);

  // ---- layer 2 (skip + biases written straight into d_out) ----
  k_init_md<<<g_init, 256, 0, stream>>>(mbuf, den, NH4);
  k_gemm3<<<g_gemm, 128, 0, stream>>>(hbuf, W2s, W2d, Wl2, bl2, b2, xs, xd, out);
  k_rowdot<<<g_row, 256, 0, stream>>>(xs, xd, at2s, at2d, a_s, a_d);
  k_edge_logit<<<g_eh, 256, 0, stream>>>(src, dst, a_s, a_d, ebuf, mbuf);
  k_edge_exp<<<g_eh, 256, 0, stream>>>(dst, ebuf, mbuf, den);
  k_edge_aggr<<<g_agg, 256, 0, stream>>>(src, dst, ebuf, den, xs, out);
}
